// Hybrid_emb_17351667876345
// MI455X (gfx1250) — matrix operand order repaired and checked by reading, not compiled, not run on hardware
//
#include <hip/hip_runtime.h>
#include <hip/hip_fp16.h>

typedef __attribute__((ext_vector_type(16))) _Float16 v16h;
typedef __attribute__((ext_vector_type(8)))  _Float16 v8h;
typedef __attribute__((ext_vector_type(8)))  float    v8f;

#define HDIM  2048
#define NHEAD 32
#define HEADD 64
#define BATCH 32
#define SEQ   1024
#define LQ    64
#define NEGMAX (-3.4028234663852886e38f)

static __device__ __forceinline__ v8f wmma16(v16h a, v16h b, v8f c) {
  // D = A(16x32 f16) * B(32x16 f16) + C(16x16 f32)
  return __builtin_amdgcn_wmma_f32_16x16x32_f16(false, a, false, b, (short)0, c, false, false);
}

static __device__ __forceinline__ v16h pack16(float4 a0, float4 a1, float4 a2, float4 a3) {
  v16h r;
  r[0]=(_Float16)a0.x;  r[1]=(_Float16)a0.y;  r[2]=(_Float16)a0.z;  r[3]=(_Float16)a0.w;
  r[4]=(_Float16)a1.x;  r[5]=(_Float16)a1.y;  r[6]=(_Float16)a1.z;  r[7]=(_Float16)a1.w;
  r[8]=(_Float16)a2.x;  r[9]=(_Float16)a2.y;  r[10]=(_Float16)a2.z; r[11]=(_Float16)a2.w;
  r[12]=(_Float16)a3.x; r[13]=(_Float16)a3.y; r[14]=(_Float16)a3.z; r[15]=(_Float16)a3.w;
  return r;
}

// ---------------- elementwise f32 -> f16 ----------------
__global__ void f32_to_f16_kernel(const float* __restrict__ src,
                                  _Float16* __restrict__ dst, int n) {
  int i = blockIdx.x * blockDim.x + threadIdx.x;
  int stride = gridDim.x * blockDim.x;
  for (; i < n; i += stride) dst[i] = (_Float16)src[i];
}

// ---------------- per-row rsqrt(mean(x^2)+eps) ----------------
__global__ void rms_scale_kernel(const float* __restrict__ x, float* __restrict__ rs) {
  __shared__ float red[256];
  const int row = blockIdx.x;
  const float* p = x + (size_t)row * HDIM;
  float s = 0.f;
  for (int i = threadIdx.x; i < HDIM; i += 256) { float v = p[i]; s += v * v; }
  red[threadIdx.x] = s;
  __syncthreads();
  for (int st = 128; st > 0; st >>= 1) {
    if (threadIdx.x < st) red[threadIdx.x] += red[threadIdx.x + st];
    __syncthreads();
  }
  if (threadIdx.x == 0) rs[row] = rsqrtf(red[0] * (1.0f / HDIM) + 1e-6f);
}

// ---------------- fused RMSNorm + dual K/V projection ----------------
// A = mol_rep [32768, 2048] f32 (normalized on the fly); W row-major [2048,2048] f16.
// k written row-major f16 [B*S, H]; v written TRANSPOSED f16 [B, NH, HD, S].
__global__ __launch_bounds__(256) void kv_proj_kernel(
    const float* __restrict__ mol, const float* __restrict__ rsc,
    const float* __restrict__ gw,
    const _Float16* __restrict__ wk, const _Float16* __restrict__ wv,
    _Float16* __restrict__ kout, _Float16* __restrict__ vtout) {
  const int lane = threadIdx.x & 31;
  const int w    = threadIdx.x >> 5;
  const int lm   = lane & 15, hi = lane >> 4;
  const int row0 = blockIdx.x * 32 + (w & 1) * 16;
  const int n0   = blockIdx.y * 256 + (w >> 1) * 64;
  const int row  = row0 + lm;
  const float rs = rsc[row];
  const float* arow = mol + (size_t)row * HDIM;

  v8f z = {};
  v8f ak[4] = {z, z, z, z};
  v8f av[4] = {z, z, z, z};

  for (int k0 = 0; k0 < HDIM; k0 += 32) {
    const int c0 = k0 + 16 * hi;
    const int c1 = k0 + 16 * hi + 8;
    float4 x0 = *(const float4*)(arow + c0);
    float4 x1 = *(const float4*)(arow + c0 + 4);
    float4 x2 = *(const float4*)(arow + c1);
    float4 x3 = *(const float4*)(arow + c1 + 4);
    const float4 g0 = *(const float4*)(gw + c0);
    const float4 g1 = *(const float4*)(gw + c0 + 4);
    const float4 g2 = *(const float4*)(gw + c1);
    const float4 g3 = *(const float4*)(gw + c1 + 4);
    x0.x *= rs * g0.x; x0.y *= rs * g0.y; x0.z *= rs * g0.z; x0.w *= rs * g0.w;
    x1.x *= rs * g1.x; x1.y *= rs * g1.y; x1.z *= rs * g1.z; x1.w *= rs * g1.w;
    x2.x *= rs * g2.x; x2.y *= rs * g2.y; x2.z *= rs * g2.z; x2.w *= rs * g2.w;
    x3.x *= rs * g3.x; x3.y *= rs * g3.y; x3.z *= rs * g3.z; x3.w *= rs * g3.w;
    const v16h a = pack16(x0, x1, x2, x3);
#pragma unroll
    for (int nt = 0; nt < 4; ++nt) {
      const int n = n0 + nt * 16 + lm;
      const v16h bk = *(const v16h*)(wk + (size_t)n * HDIM + k0 + 16 * hi);
      const v16h bv = *(const v16h*)(wv + (size_t)n * HDIM + k0 + 16 * hi);
      ak[nt] = wmma16(a, bk, ak[nt]);
      av[nt] = wmma16(a, bv, av[nt]);
    }
  }

  const int b     = row0 >> 10;               // row0 / SEQ
  const int sbase = (row0 & (SEQ - 1)) + 8 * hi;
#pragma unroll
  for (int nt = 0; nt < 4; ++nt) {
    const int col = n0 + nt * 16 + lm;
#pragma unroll
    for (int r = 0; r < 8; ++r)
      kout[(size_t)(row0 + r + 8 * hi) * HDIM + col] = (_Float16)ak[nt][r];
    const int h = col >> 6, d = col & 63;
    union { _Float16 hv[8]; int4 q; } pk;
#pragma unroll
    for (int r = 0; r < 8; ++r) pk.hv[r] = (_Float16)av[nt][r];
    *(int4*)(vtout + ((size_t)(b * NHEAD + h) * HEADD + d) * SEQ + sbase) = pk.q;
  }
}

// ---------------- generic GEMM: C = A(f32) * W(f16)^T ----------------
template <bool OUT_F16>
__global__ __launch_bounds__(256) void gemm_kernel(
    const float* __restrict__ A, const _Float16* __restrict__ W,
    void* __restrict__ Cout) {
  const int lane = threadIdx.x & 31;
  const int w    = threadIdx.x >> 5;
  const int lm   = lane & 15, hi = lane >> 4;
  const int row0 = blockIdx.x * 32 + (w & 1) * 16;
  const int n0   = blockIdx.y * 256 + (w >> 1) * 64;
  const float* arow = A + (size_t)(row0 + lm) * HDIM;

  v8f z = {};
  v8f acc[4] = {z, z, z, z};

  for (int k0 = 0; k0 < HDIM; k0 += 32) {
    const int c0 = k0 + 16 * hi;
    const int c1 = k0 + 16 * hi + 8;
    const float4 x0 = *(const float4*)(arow + c0);
    const float4 x1 = *(const float4*)(arow + c0 + 4);
    const float4 x2 = *(const float4*)(arow + c1);
    const float4 x3 = *(const float4*)(arow + c1 + 4);
    const v16h a = pack16(x0, x1, x2, x3);
#pragma unroll
    for (int nt = 0; nt < 4; ++nt) {
      const int n = n0 + nt * 16 + lm;
      const v16h b = *(const v16h*)(W + (size_t)n * HDIM + k0 + 16 * hi);
      acc[nt] = wmma16(a, b, acc[nt]);
    }
  }
#pragma unroll
  for (int nt = 0; nt < 4; ++nt) {
    const int col = n0 + nt * 16 + lm;
#pragma unroll
    for (int r = 0; r < 8; ++r) {
      const size_t idx = (size_t)(row0 + r + 8 * hi) * HDIM + col;
      if constexpr (OUT_F16) ((_Float16*)Cout)[idx] = (_Float16)acc[nt][r];
      else                   ((float*)Cout)[idx] = acc[nt][r];
    }
  }
}

// ---------------- flash-style cross attention per (b, h) ----------------
__global__ __launch_bounds__(128) void attn_kernel(
    const _Float16* __restrict__ q16, const _Float16* __restrict__ k16,
    const _Float16* __restrict__ vt16, const unsigned char* __restrict__ msk,
    float* __restrict__ oatt) {
  __shared__ alignas(32) _Float16 plds[4][16][32];
  const int h = blockIdx.x, b = blockIdx.y;
  const int lane = threadIdx.x & 31, w = threadIdx.x >> 5;
  const int lm = lane & 15, hi = lane >> 4;
  const int l0 = w * 16;

  // preload Q as two A tiles (K = 0..31, 32..63)
  const _Float16* qrow = q16 + (size_t)(l0 + lm) * HDIM + h * HEADD;
  v16h aq[2];
#pragma unroll
  for (int kk = 0; kk < 2; ++kk) {
    const v8h lo = *(const v8h*)(qrow + kk * 32 + 16 * hi);
    const v8h hh = *(const v8h*)(qrow + kk * 32 + 16 * hi + 8);
#pragma unroll
    for (int j = 0; j < 8; ++j) { aq[kk][j] = lo[j]; aq[kk][8 + j] = hh[j]; }
  }

  v8f z = {};
  v8f oacc[4] = {z, z, z, z};
  float mrun[8], rrun[8];
#pragma unroll
  for (int r = 0; r < 8; ++r) { mrun[r] = NEGMAX; rrun[r] = 0.f; }

  const _Float16* kbase = k16 + (size_t)b * SEQ * HDIM + h * HEADD;
  const _Float16* vbase = vt16 + (size_t)(b * NHEAD + h) * HEADD * SEQ;
  const size_t mbase = (size_t)b * LQ * SEQ;

  for (int s0 = 0; s0 < SEQ; s0 += 32) {
    // ---- scores tile [16 x 32] ----
    v8f sc[2];
#pragma unroll
    for (int t = 0; t < 2; ++t) {
      v8f a = z;
      const int s = s0 + 16 * t + lm;
      const _Float16* krow = kbase + (size_t)s * HDIM;
#pragma unroll
      for (int kk = 0; kk < 2; ++kk) {
        const v16h bk = *(const v16h*)(krow + kk * 32 + 16 * hi);
        a = wmma16(aq[kk], bk, a);
      }
      a = a * 0.125f;  // 1/sqrt(HD)
#pragma unroll
      for (int r = 0; r < 8; ++r) {
        const int l = l0 + r + 8 * hi;
        if (msk[mbase + (size_t)l * SEQ + s]) a[r] = NEGMAX;
      }
      sc[t] = a;
    }
    // ---- online softmax update ----
#pragma unroll
    for (int r = 0; r < 8; ++r) {
      float mx = fmaxf(sc[0][r], sc[1][r]);
#pragma unroll
      for (int off = 8; off >= 1; off >>= 1) mx = fmaxf(mx, __shfl_xor(mx, off, 16));
      const float mnew = fmaxf(mrun[r], mx);
      const float corr = __expf(mrun[r] - mnew);
      const float p0 = __expf(sc[0][r] - mnew);
      const float p1 = __expf(sc[1][r] - mnew);
      float rsum = p0 + p1;
#pragma unroll
      for (int off = 8; off >= 1; off >>= 1) rsum += __shfl_xor(rsum, off, 16);
      rrun[r] = rrun[r] * corr + rsum;
      mrun[r] = mnew;
#pragma unroll
      for (int nt = 0; nt < 4; ++nt) oacc[nt][r] *= corr;
      plds[w][r + 8 * hi][lm]      = (_Float16)p0;
      plds[w][r + 8 * hi][16 + lm] = (_Float16)p1;
    }
    __syncthreads();
    // ---- read P back in A-layout ----
    v16h ap;
    {
      const v8h lo = *(const v8h*)&plds[w][lm][16 * hi];
      const v8h hh = *(const v8h*)&plds[w][lm][16 * hi + 8];
#pragma unroll
      for (int j = 0; j < 8; ++j) { ap[j] = lo[j]; ap[8 + j] = hh[j]; }
    }
    __syncthreads();
    // ---- O += P * V (V transposed in memory -> contiguous B loads) ----
#pragma unroll
    for (int nt = 0; nt < 4; ++nt) {
      const int d = nt * 16 + lm;
      const v16h bv = *(const v16h*)(vbase + (size_t)d * SEQ + s0 + 16 * hi);
      oacc[nt] = wmma16(ap, bv, oacc[nt]);
    }
  }
  // ---- epilogue: normalize and store [B, L, H] f32 ----
#pragma unroll
  for (int r = 0; r < 8; ++r) {
    const float inv = 1.0f / rrun[r];
    const int l = l0 + r + 8 * hi;
#pragma unroll
    for (int nt = 0; nt < 4; ++nt)
      oatt[((size_t)b * LQ + l) * HDIM + h * HEADD + nt * 16 + lm] = oacc[nt][r] * inv;
  }
}

extern "C" void kernel_launch(void* const* d_in, const int* in_sizes, int n_in,
                              void* d_out, int out_size, void* d_ws, size_t ws_size,
                              hipStream_t stream) {
  (void)in_sizes; (void)n_in; (void)out_size; (void)ws_size;
  const float* mol = (const float*)d_in[0];
  const float* adp = (const float*)d_in[1];
  const float* gw  = (const float*)d_in[2];
  const float* wq  = (const float*)d_in[3];
  const float* wk  = (const float*)d_in[4];
  const float* wv  = (const float*)d_in[5];
  const float* wo  = (const float*)d_in[6];
  const unsigned char* msk = (const unsigned char*)d_in[7];

  char* ws = (char*)d_ws;
  _Float16* wq16 = (_Float16*)(ws + 0);
  _Float16* wk16 = (_Float16*)(ws + (8ull  << 20));
  _Float16* wv16 = (_Float16*)(ws + (16ull << 20));
  _Float16* wo16 = (_Float16*)(ws + (24ull << 20));
  float*    rsc  = (float*)   (ws + (32ull << 20));
  _Float16* q16  = (_Float16*)(ws + (32ull << 20) + (1ull << 18));
  _Float16* k16  = (_Float16*)(ws + (33ull << 20));
  _Float16* vt16 = (_Float16*)(ws + (33ull << 20) + (128ull << 20));
  float*    att  = (float*)   (ws + (33ull << 20) + (256ull << 20));
  // total workspace used: ~305 MB

  const int WN = HDIM * HDIM;
  f32_to_f16_kernel<<<1024, 256, 0, stream>>>(wq, wq16, WN);
  f32_to_f16_kernel<<<1024, 256, 0, stream>>>(wk, wk16, WN);
  f32_to_f16_kernel<<<1024, 256, 0, stream>>>(wv, wv16, WN);
  f32_to_f16_kernel<<<1024, 256, 0, stream>>>(wo, wo16, WN);

  rms_scale_kernel<<<BATCH * SEQ, 256, 0, stream>>>(mol, rsc);

  // q = adaptor @ wq^T : M=64
  gemm_kernel<true><<<dim3(2, 8), 256, 0, stream>>>(adp, wq16, (void*)q16);

  // fused rmsnorm + k/v projections : M=32768
  kv_proj_kernel<<<dim3(1024, 8), 256, 0, stream>>>(mol, rsc, gw, wk16, wv16, k16, vt16);

  // attention per (b, h)
  attn_kernel<<<dim3(NHEAD, BATCH), 128, 0, stream>>>(q16, k16, vt16, msk, att);

  // o_proj : M = B*L = 2048, f32 output
  gemm_kernel<false><<<dim3((BATCH * LQ) / 32, 8), 256, 0, stream>>>(att, wo16, d_out);
}
